// Gemma4AudioAttention_70626442216133
// MI455X (gfx1250) — matrix operand order repaired and checked by reading, not compiled, not run on hardware
//
#include <hip/hip_runtime.h>
#include <stdint.h>

// ---------------------------------------------------------------------------
// Gemma4 audio attention for MI455X (gfx1250, wave32, WMMA).
// bf16 WMMA for all heavy matmuls, f32 accumulation, f32 softmax.
// Tile staging uses CDNA5 async global->LDS DMA (ASYNCcnt) to avoid VGPR
// staging (previous revision spilled to scratch).
// ---------------------------------------------------------------------------

#define Hn     12
#define Dd     128
#define Ee     1536
#define CHUNK  12
#define CTX    24
#define Bb     8
#define Ss     3072
#define NBk    256            // Ss / CHUNK
#define Mtot   (Bb * Ss)      // 24576

// Q_SCALE = D^-0.5 / ln(2);  K_SCALE = ln(1+e)/ln(2)
#define QSCALE 0.12751744f
#define KSCALE 1.8946357f

typedef __attribute__((ext_vector_type(16))) __bf16 v16bf;
typedef __attribute__((ext_vector_type(8)))  float  v8f;

union FragBF { uint4 u[2]; v16bf v; };

__device__ __forceinline__ v16bf frag_ld(const uint16_t* p0, const uint16_t* p1) {
  FragBF f;
  f.u[0] = *reinterpret_cast<const uint4*>(p0);
  f.u[1] = *reinterpret_cast<const uint4*>(p1);
  return f.v;
}

__device__ __forceinline__ uint16_t f32_to_bf16(float f) {
  union { float f; uint32_t u; } c; c.f = f;
  uint32_t u = c.u;
  u += 0x7FFFu + ((u >> 16) & 1u);     // round-to-nearest-even
  return (uint16_t)(u >> 16);
}

// Async global -> LDS 16B DMA (per-lane), tracked by ASYNCcnt.
// lds: LDS byte address (low 32 bits of generic shared pointer),
// goff: 32-bit unsigned byte offset from sgpr base.
__device__ __forceinline__ void async_ld_b128(uint32_t lds, uint32_t goff,
                                              const void* base) {
  asm volatile("global_load_async_to_lds_b128 %0, %1, %2"
               :: "v"(lds), "v"(goff), "s"(base) : "memory");
}
__device__ __forceinline__ void wait_async0() {
  asm volatile("s_wait_asynccnt 0x0" ::: "memory");
}
__device__ __forceinline__ uint32_t lds_addr(const void* p) {
  return (uint32_t)(uintptr_t)p;       // LDS aperture: low 32 bits = LDS offset
}

// ---------------------------------------------------------------------------
// Elementwise f32 -> bf16 conversion (feeds the WMMA pipeline)
// ---------------------------------------------------------------------------
__global__ void cvt_bf16_kernel(const float* __restrict__ in,
                                uint16_t* __restrict__ out, size_t n) {
  size_t i = (size_t)blockIdx.x * blockDim.x + threadIdx.x;
  size_t stride = (size_t)gridDim.x * blockDim.x;
  for (; i < n; i += stride) out[i] = f32_to_bf16(in[i]);
}

// ---------------------------------------------------------------------------
// rel_k = position_embeddings @ w_rel.T  (24 x 1536 output; tiny, naive f32)
// ---------------------------------------------------------------------------
__global__ void rel_proj_kernel(const float* __restrict__ pos,
                                const float* __restrict__ wrel,
                                uint16_t* __restrict__ relB) {
  int idx = blockIdx.x * blockDim.x + threadIdx.x;   // p*E + n
  if (idx >= CTX * Ee) return;
  int p = idx / Ee, n = idx % Ee;
  const float* a = pos  + (size_t)p * Ee;
  const float* w = wrel + (size_t)n * Ee;
  float s = 0.f;
  for (int k = 0; k < Ee; ++k) s = fmaf(a[k], w[k], s);
  relB[idx] = f32_to_bf16(s);
}

// ---------------------------------------------------------------------------
// WMMA GEMM:  out[m,n] = sum_k A[m,k] * W[n,k]   (torch Linear, W row-major)
// Block tile 128x128, 8 waves (4x2), wave tile 32x64, K-step 32.
// LDS ping-pong double buffering via async global->LDS DMA + L2 prefetch.
// mode: 0=q (per-dim softplus scale, bf16 out), 1=k (KSCALE, bf16),
//       2=v (bf16), 3=post (f32 out to d_out)
// ---------------------------------------------------------------------------
#define LDST 40     // LDS row stride in bf16 elems (80B: conflict-free, 16B mult)

__global__ __launch_bounds__(256, 1)
void wmma_gemm_kernel(const uint16_t* __restrict__ A,
                      const uint16_t* __restrict__ Wb,
                      void* __restrict__ out,
                      const float* __restrict__ pds,
                      int mode)
{
  __shared__ __align__(16) uint16_t As[2][128 * LDST];
  __shared__ __align__(16) uint16_t Bs[2][128 * LDST];

  const int tid  = threadIdx.x;
  const int lane = tid & 31;
  const int wave = tid >> 5;
  const int wm   = wave & 3;                  // M offset 32*wm
  const int wn   = wave >> 2;                 // N offset 64*wn
  const int nbn  = Ee / 128;                  // 12 N-blocks
  const int mbase = (blockIdx.x / nbn) * 128;
  const int nbase = (blockIdx.x % nbn) * 128;
  const int ksteps = Ee / 32;                 // 48

  const int hl = lane >> 4;                   // lane half (K-half select)
  const int lr = lane & 15;                   // row/col within tile

  // Per-thread tile chunks: 512 x 16B per matrix, 2 per thread.
  const int row0 = tid >> 2;                  // chunk row for c = tid
  const int off0 = (tid & 3) * 8;             // bf16-elem offset within row

  auto async_tile = [&](int ks, int buf) {
    const int kb = ks * 32;
#pragma unroll
    for (int i = 0; i < 2; ++i) {
      const int c = tid + i * 256;
      const int row = c >> 2, off = (c & 3) * 8;
      const uint32_t l = row * LDST + off;
      async_ld_b128(lds_addr(&As[buf][l]),
                    (uint32_t)((((size_t)(mbase + row)) * Ee + kb + off) * 2), A);
      async_ld_b128(lds_addr(&Bs[buf][l]),
                    (uint32_t)((((size_t)(nbase + row)) * Ee + kb + off) * 2), Wb);
    }
  };

  v8f acc[2][4];
  v8f z = {};
#pragma unroll
  for (int i = 0; i < 2; ++i)
#pragma unroll
    for (int j = 0; j < 4; ++j) acc[i][j] = z;

  async_tile(0, 0);
  wait_async0();
  __syncthreads();

  for (int ks = 0; ks < ksteps; ++ks) {
    const int cur = ks & 1;
    if (ks + 1 < ksteps) async_tile(ks + 1, cur ^ 1);
    if (ks + 2 < ksteps) {                    // L2 prefetch (global_prefetch_b8)
      const int kb2 = (ks + 2) * 32;
      __builtin_prefetch(A  + (size_t)(mbase + row0) * Ee + kb2 + off0, 0, 1);
      __builtin_prefetch(Wb + (size_t)(nbase + row0) * Ee + kb2 + off0, 0, 1);
    }

    v16bf af[2], bfr[4];
#pragma unroll
    for (int mi = 0; mi < 2; ++mi) {
      const uint16_t* base = &As[cur][(wm * 32 + mi * 16 + lr) * LDST];
      af[mi] = frag_ld(base + 16 * hl, base + 16 * hl + 8);
    }
#pragma unroll
    for (int ni = 0; ni < 4; ++ni) {
      const uint16_t* base = &Bs[cur][(wn * 64 + ni * 16 + lr) * LDST];
      bfr[ni] = frag_ld(base + 16 * hl, base + 16 * hl + 8);
    }
#pragma unroll
    for (int mi = 0; mi < 2; ++mi)
#pragma unroll
      for (int ni = 0; ni < 4; ++ni)
        acc[mi][ni] = __builtin_amdgcn_wmma_f32_16x16x32_bf16(
            false, af[mi], false, bfr[ni], (short)0, acc[mi][ni], false, false);

    wait_async0();                            // next tile fully in LDS
    __syncthreads();
  }

  // ---- epilogue: C layout = VGPR r -> M = r + 8*(lane/16), N = lane%16
#pragma unroll
  for (int mi = 0; mi < 2; ++mi) {
#pragma unroll
    for (int ni = 0; ni < 4; ++ni) {
      const int col = nbase + wn * 64 + ni * 16 + lr;
      float scale = 1.0f;
      if (mode == 0) {
        float x  = pds[col & (Dd - 1)];
        float sp = (x > 20.f) ? x : log1pf(expf(x));   // softplus
        scale = QSCALE * sp;
      } else if (mode == 1) {
        scale = KSCALE;
      }
#pragma unroll
      for (int r = 0; r < 8; ++r) {
        const int mrow = mbase + wm * 32 + mi * 16 + r + 8 * hl;
        const float v = acc[mi][ni][r] * scale;
        if (mode == 3) {
          reinterpret_cast<float*>(out)[(size_t)mrow * Ee + col] = v;
        } else {
          reinterpret_cast<uint16_t*>(out)[(size_t)mrow * Ee + col] = f32_to_bf16(v);
        }
      }
    }
  }
}

// ---------------------------------------------------------------------------
// Chunked local attention. One wave per (b, h, chunk-block).
//   logits = Q Kctx^T + rel_shift(Q Rel^T); softcap; softmax; out = P Vctx
// Q/K/rel tiles staged with async global->LDS DMA; pad regions zeroed via DS.
// ---------------------------------------------------------------------------
#define QST 136   // 272B row stride (bank-conflict free, 16B mult)
#define KST 136
#define VST 40
#define AST 40
#define FST 34

__global__ __launch_bounds__(32)
void attn_kernel(const uint16_t* __restrict__ qB, const uint16_t* __restrict__ kB,
                 const uint16_t* __restrict__ vB, const uint16_t* __restrict__ relB,
                 uint16_t* __restrict__ ctxB, float* __restrict__ attnW)
{
  __shared__ __align__(16) uint16_t qS[16 * QST];
  __shared__ __align__(16) uint16_t kS[32 * KST];
  __shared__ __align__(16) uint16_t rS[32 * KST];
  __shared__ __align__(16) uint16_t vT[128 * VST];   // v transposed: [d][j]
  __shared__ __align__(16) uint16_t aS[16 * AST];    // attn probs bf16
  __shared__ float acS[16 * FST];
  __shared__ float bdS[16 * FST];

  const int lane = threadIdx.x;
  const int wg = blockIdx.x;              // ((b*H + h)*NB + nb)
  const int nb = wg % NBk;
  const int h  = (wg / NBk) % Hn;
  const int b  = wg / (NBk * Hn);
  const int hl = lane >> 4, lr = lane & 15;
  const uint4 zz = {0, 0, 0, 0};

  // ---- Q tile: 16 rows x 128 (rows >= CHUNK zero)
#pragma unroll
  for (int i = 0; i < 8; ++i) {
    int c = lane + i * 32;                 // 256 chunks of 16B
    int row = c >> 4, off = (c & 15) * 8;
    if (row < CHUNK) {
      size_t s = (size_t)b * Ss + (size_t)nb * CHUNK + row;
      async_ld_b128(lds_addr(&qS[row * QST + off]),
                    (uint32_t)((s * Ee + h * Dd + off) * 2), qB);
    } else {
      *reinterpret_cast<uint4*>(&qS[row * QST + off]) = zz;
    }
  }
  // ---- K ctx: 32 rows x 128, ctx j -> seq s = nb*CHUNK - 12 + j (zero pad)
#pragma unroll
  for (int i = 0; i < 16; ++i) {
    int c = lane + i * 32;                 // 512 chunks
    int j = c >> 4, off = (c & 15) * 8;
    int s = nb * CHUNK - 12 + j;
    if (j < CTX && s >= 0 && s < Ss) {
      async_ld_b128(lds_addr(&kS[j * KST + off]),
                    (uint32_t)((((size_t)b * Ss + s) * Ee + h * Dd + off) * 2), kB);
    } else {
      *reinterpret_cast<uint4*>(&kS[j * KST + off]) = zz;
    }
  }
  // ---- Rel: rows 0..23 from relB[p, h*D + d]
#pragma unroll
  for (int i = 0; i < 16; ++i) {
    int c = lane + i * 32;
    int p = c >> 4, off = (c & 15) * 8;
    if (p < CTX) {
      async_ld_b128(lds_addr(&rS[p * KST + off]),
                    (uint32_t)(((size_t)p * Ee + h * Dd + off) * 2), relB);
    } else {
      *reinterpret_cast<uint4*>(&rS[p * KST + off]) = zz;
    }
  }
  // ---- V transposed into LDS: vT[d][j]
  for (int idx = lane; idx < 128 * 32; idx += 32) {
    int j = idx >> 7, d = idx & 127;
    uint16_t val = 0;
    int s = nb * CHUNK - 12 + j;
    if (j < CTX && s >= 0 && s < Ss)
      val = vB[((size_t)b * Ss + s) * Ee + h * Dd + d];
    vT[d * VST + j] = val;
  }
  for (int i = lane; i < 16 * AST; i += 32) aS[i] = 0;
  wait_async0();
  __syncthreads();

  // ---- content (Q K^T) and position (Q Rel^T) logits via WMMA
  v16bf aq[4];
#pragma unroll
  for (int ks = 0; ks < 4; ++ks) {
    const uint16_t* base = &qS[lr * QST + ks * 32];
    aq[ks] = frag_ld(base + 16 * hl, base + 16 * hl + 8);
  }
  v8f z = {};
  v8f accA[2] = {z, z}, accB[2] = {z, z};
#pragma unroll
  for (int nt = 0; nt < 2; ++nt) {
#pragma unroll
    for (int ks = 0; ks < 4; ++ks) {
      const uint16_t* bk = &kS[(nt * 16 + lr) * KST + ks * 32];
      v16bf kf = frag_ld(bk + 16 * hl, bk + 16 * hl + 8);
      accA[nt] = __builtin_amdgcn_wmma_f32_16x16x32_bf16(
          false, aq[ks], false, kf, (short)0, accA[nt], false, false);
      const uint16_t* br = &rS[(nt * 16 + lr) * KST + ks * 32];
      v16bf rf = frag_ld(br + 16 * hl, br + 16 * hl + 8);
      accB[nt] = __builtin_amdgcn_wmma_f32_16x16x32_bf16(
          false, aq[ks], false, rf, (short)0, accB[nt], false, false);
    }
  }
#pragma unroll
  for (int nt = 0; nt < 2; ++nt)
#pragma unroll
    for (int r = 0; r < 8; ++r) {
      int m = r + 8 * hl, n = nt * 16 + lr;
      acS[m * FST + n] = accA[nt][r];
      bdS[m * FST + n] = accB[nt][r];
    }
  __syncthreads();

  // ---- rel-shift + softcap + softmax (lane c handles query row c)
  if (lane < CHUNK) {
    const int c = lane;
    float lg[CTX];
    float mx = -1e30f;
#pragma unroll
    for (int j = 0; j < CTX; ++j) {
      // exact reference _rel_shift semantics (incl. wrap rows)
      int sidx = c * CTX + j;
      int sc = sidx / (CTX + 1);
      int sp = sidx % (CTX + 1);
      float bd = (sp == CTX) ? 0.f : bdS[sc * FST + sp];
      float l = acS[c * FST + j] + bd;
      l = tanhf(l * 0.02f) * 50.0f;         // softcap
      lg[j] = l;
      mx = fmaxf(mx, l);
    }
    float sum = 0.f;
#pragma unroll
    for (int j = 0; j < CTX; ++j) { lg[j] = expf(lg[j] - mx); sum += lg[j]; }
    float inv = 1.0f / sum;
    float* aw = attnW + (size_t)wg * (CHUNK * CTX) + c * CTX;
#pragma unroll
    for (int j = 0; j < CTX; ++j) {
      float w = lg[j] * inv;
      aw[j] = w;
      aS[c * AST + j] = f32_to_bf16(w);
    }
  }
  __syncthreads();

  // ---- out = P (16x32) @ Vctx (32x128) via 8 WMMAs
  const uint16_t* ab = &aS[lr * AST];
  v16bf pf = frag_ld(ab + 16 * hl, ab + 16 * hl + 8);
  const size_t outRow = (size_t)b * Ss + (size_t)nb * CHUNK;
#pragma unroll
  for (int nt = 0; nt < 8; ++nt) {
    const uint16_t* vb = &vT[(nt * 16 + lr) * VST];
    v16bf vf = frag_ld(vb + 16 * hl, vb + 16 * hl + 8);
    v8f acc = z;
    acc = __builtin_amdgcn_wmma_f32_16x16x32_bf16(
        false, pf, false, vf, (short)0, acc, false, false);
#pragma unroll
    for (int r = 0; r < 8; ++r) {
      int m = r + 8 * hl;
      if (m < CHUNK) {
        int d = nt * 16 + lr;
        ctxB[(outRow + m) * Ee + h * Dd + d] = f32_to_bf16(acc[r]);
      }
    }
  }
}

// ---------------------------------------------------------------------------
// Launch
// ---------------------------------------------------------------------------
extern "C" void kernel_launch(void* const* d_in, const int* in_sizes, int n_in,
                              void* d_out, int out_size, void* d_ws, size_t ws_size,
                              hipStream_t stream) {
  (void)in_sizes; (void)n_in; (void)out_size; (void)ws_size;
  const float* hidden = (const float*)d_in[0];
  const float* pos    = (const float*)d_in[1];
  const float* w_q    = (const float*)d_in[2];
  const float* w_k    = (const float*)d_in[3];
  const float* w_v    = (const float*)d_in[4];
  const float* w_post = (const float*)d_in[5];
  const float* w_rel  = (const float*)d_in[6];
  const float* pds    = (const float*)d_in[7];

  const size_t nHid = (size_t)Mtot * Ee;    // 37,748,736
  const size_t nW   = (size_t)Ee * Ee;      // 2,359,296

  uint16_t* p = (uint16_t*)d_ws;
  uint16_t* hiddenB = p; p += nHid;         // reused as ctxB after v-GEMM
  uint16_t* wqB  = p; p += nW;
  uint16_t* wkB  = p; p += nW;
  uint16_t* wvB  = p; p += nW;
  uint16_t* wpB  = p; p += nW;
  uint16_t* relB = p; p += (size_t)CTX * Ee;
  uint16_t* qB   = p; p += nHid;
  uint16_t* kB   = p; p += nHid;
  uint16_t* vB   = p; p += nHid;
  uint16_t* ctxB = hiddenB;

  float* outMain = (float*)d_out;
  float* attnW   = outMain + nHid;

  // 1) bf16 conversions
  cvt_bf16_kernel<<<4096, 256, 0, stream>>>(hidden, hiddenB, nHid);
  cvt_bf16_kernel<<<2048, 256, 0, stream>>>(w_q,    wqB, nW);
  cvt_bf16_kernel<<<2048, 256, 0, stream>>>(w_k,    wkB, nW);
  cvt_bf16_kernel<<<2048, 256, 0, stream>>>(w_v,    wvB, nW);
  cvt_bf16_kernel<<<2048, 256, 0, stream>>>(w_post, wpB, nW);

  // 2) rel_k projection (tiny)
  rel_proj_kernel<<<(CTX * Ee + 255) / 256, 256, 0, stream>>>(pos, w_rel, relB);

  // 3) q/k/v projections (WMMA GEMMs)
  const int gemmGrid = (Mtot / 128) * (Ee / 128);   // 192 * 12 = 2304
  wmma_gemm_kernel<<<gemmGrid, 256, 0, stream>>>(hiddenB, wqB, qB, pds, 0);
  wmma_gemm_kernel<<<gemmGrid, 256, 0, stream>>>(hiddenB, wkB, kB, pds, 1);
  wmma_gemm_kernel<<<gemmGrid, 256, 0, stream>>>(hiddenB, wvB, vB, pds, 2);

  // 4) chunked local attention (one wave per (b,h,block))
  attn_kernel<<<Bb * Hn * NBk, 32, 0, stream>>>(qB, kB, vB, relB, ctxB, attnW);

  // 5) post projection -> fp32 d_out
  wmma_gemm_kernel<<<gemmGrid, 256, 0, stream>>>(ctxB, wpB, outMain, pds, 3);
}
